// NanoGPT_7645041787218
// MI455X (gfx1250) — matrix operand order repaired and checked by reading, not compiled, not run on hardware
//
#include <hip/hip_runtime.h>

#define DEV __device__ __forceinline__

typedef __bf16 bf16_t;
typedef __attribute__((ext_vector_type(16))) __bf16 v16bf;
typedef __attribute__((ext_vector_type(8)))  __bf16 v8bf;
typedef __attribute__((ext_vector_type(8)))  float  v8f;

constexpr int kB  = 2, kT = 1024, kC = 768, kH = 12, kL = 12, kV = 50257, kD = 64;
constexpr int kBT = kB * kT;   // 2048
constexpr int k3C = 3 * kC;    // 2304
constexpr int k4C = 4 * kC;    // 3072
constexpr int kBH = kB * kH;   // 24

enum { MODE_F32 = 0, MODE_RESID = 1, MODE_GELU = 2 };

// ---------------------------------------------------------------- WMMA helpers

DEV v8f wmma_bf16(v16bf a, v16bf b, v8f c) {
  // D = A(16x32 bf16) * B(32x16 bf16) + C(16x16 f32)
  return __builtin_amdgcn_wmma_f32_16x16x32_bf16(false, a, false, b, (short)0, c,
                                                 false, false);
}

// A fragment: row-major A[M,K].  Lane m(<16): K=16h..16h+15 contiguous.
DEV v16bf load_a_frag(const bf16_t* __restrict__ A, int lda, int row0, int k0, int lane) {
  int m = lane & 15, half = lane >> 4;
  const bf16_t* p = A + (size_t)(row0 + m) * lda + k0 + 16 * half;
  return *(const v16bf*)p;           // 32B
}

// B fragment from transposed weights Bt[N,K]: lane n(<16) holds col n, K=0..15;
// lane n+16 holds K=16..31 -> one contiguous 32B load per lane.
DEV v16bf load_b_frag(const bf16_t* __restrict__ Bt, int ldb, int col0, int k0,
                      int lane, int nmax) {
  int n = lane & 15, half = lane >> 4;
  int col = col0 + n; if (col > nmax) col = nmax;  // N-tail clamp (store guarded)
  return *(const v16bf*)(Bt + (size_t)col * ldb + k0 + 16 * half);
}

DEV float gelu_tanh(float u) {
  return 0.5f * u * (1.0f + tanhf(0.7978845608028654f * (u + 0.044715f * u * u * u)));
}

// ---------------------------------------------------------------- generic GEMM
// C[M,N] = A[M,K](bf16) * Bt[N,K](bf16)^T + bias, epilogue per mode.
// block = 8 waves, wave tile 32x32, block tile 64(M) x 128(N).
__global__ __launch_bounds__(256)
void gemm_bf16_kernel(const bf16_t* __restrict__ A, int lda,
                      const bf16_t* __restrict__ Bt, int ldb,
                      const float* __restrict__ bias,
                      const float* __restrict__ resid,
                      float* __restrict__ outf, bf16_t* __restrict__ outb,
                      int M, int N, int K, int ldo, int mode) {
  int lane = threadIdx.x & 31;
  int wave = threadIdx.x >> 5;
  int wm = wave >> 2, wn = wave & 3;
  int row0 = blockIdx.y * 64 + wm * 32;
  int col0 = blockIdx.x * 128 + wn * 32;
  if (row0 >= M) return;
  int nmax = N - 1;

  v8f acc[2][2] = {};
  for (int k0 = 0; k0 < K; k0 += 32) {
    v16bf a0 = load_a_frag(A, lda, row0,      k0, lane);
    v16bf a1 = load_a_frag(A, lda, row0 + 16, k0, lane);
    v16bf b0 = load_b_frag(Bt, ldb, col0,      k0, lane, nmax);
    v16bf b1 = load_b_frag(Bt, ldb, col0 + 16, k0, lane, nmax);
    acc[0][0] = wmma_bf16(a0, b0, acc[0][0]);
    acc[0][1] = wmma_bf16(a0, b1, acc[0][1]);
    acc[1][0] = wmma_bf16(a1, b0, acc[1][0]);
    acc[1][1] = wmma_bf16(a1, b1, acc[1][1]);
  }

  int n = lane & 15, half = lane >> 4;
#pragma unroll
  for (int tm = 0; tm < 2; ++tm)
#pragma unroll
    for (int tn = 0; tn < 2; ++tn)
#pragma unroll
      for (int r = 0; r < 8; ++r) {
        int grow = row0 + tm * 16 + r + 8 * half;
        int gcol = col0 + tn * 16 + n;
        if (gcol >= N || grow >= M) continue;
        float v = acc[tm][tn][r];
        if (bias) v += bias[gcol];
        size_t o = (size_t)grow * ldo + gcol;
        if (mode == MODE_RESID)      outf[o] = resid[o] + v;
        else if (mode == MODE_GELU)  outb[o] = (bf16_t)gelu_tanh(v);
        else                         outf[o] = v;
      }
}

// ------------------------------------------------------- attention: Q @ K^T
// per (b,h): scores[T,T] = scale * q[T,64] @ k[T,64]^T, causal block-skip.
__global__ __launch_bounds__(256)
void attn_score_kernel(const bf16_t* __restrict__ q, const bf16_t* __restrict__ kk,
                       float* __restrict__ scores, float scale) {
  int bh = blockIdx.z;
  const bf16_t* A  = q  + (size_t)bh * kT * kD;
  const bf16_t* Bt = kk + (size_t)bh * kT * kD;   // k row-major [T,D] == [N,K]
  float* out = scores + (size_t)bh * kT * kT;
  int lane = threadIdx.x & 31;
  int wave = threadIdx.x >> 5;
  int wm = wave >> 2, wn = wave & 3;
  int row0 = blockIdx.y * 64 + wm * 32;
  int col0 = blockIdx.x * 128 + wn * 32;
  if (col0 > row0 + 31) return;  // wave tile fully above causal diagonal (uniform)

  v8f acc[2][2] = {};
#pragma unroll
  for (int k0 = 0; k0 < kD; k0 += 32) {
    v16bf a0 = load_a_frag(A, kD, row0,      k0, lane);
    v16bf a1 = load_a_frag(A, kD, row0 + 16, k0, lane);
    v16bf b0 = load_b_frag(Bt, kD, col0,      k0, lane, kT - 1);
    v16bf b1 = load_b_frag(Bt, kD, col0 + 16, k0, lane, kT - 1);
    acc[0][0] = wmma_bf16(a0, b0, acc[0][0]);
    acc[0][1] = wmma_bf16(a0, b1, acc[0][1]);
    acc[1][0] = wmma_bf16(a1, b0, acc[1][0]);
    acc[1][1] = wmma_bf16(a1, b1, acc[1][1]);
  }
  int n = lane & 15, half = lane >> 4;
#pragma unroll
  for (int tm = 0; tm < 2; ++tm)
#pragma unroll
    for (int tn = 0; tn < 2; ++tn)
#pragma unroll
      for (int r = 0; r < 8; ++r) {
        int grow = row0 + tm * 16 + r + 8 * half;
        int gcol = col0 + tn * 16 + n;
        out[(size_t)grow * kT + gcol] = acc[tm][tn][r] * scale;
      }
}

// ------------------------------------- causal softmax row -> unnormalized bf16
__global__ __launch_bounds__(256)
void softmax_kernel(const float* __restrict__ scores, bf16_t* __restrict__ att,
                    float* __restrict__ rowsum) {
  int qrow = blockIdx.x, bh = blockIdx.y;
  const float* s = scores + ((size_t)bh * kT + qrow) * kT;
  bf16_t* a = att + ((size_t)bh * kT + qrow) * kT;
  int len = qrow + 1;  // causal
  __shared__ float red[8];

  float m = -1e30f;
  for (int j = threadIdx.x; j < len; j += 256) m = fmaxf(m, s[j]);
  for (int o = 16; o > 0; o >>= 1) m = fmaxf(m, __shfl_xor(m, o, 32));
  if ((threadIdx.x & 31) == 0) red[threadIdx.x >> 5] = m;
  __syncthreads();
  float bm = red[0];
  for (int i = 1; i < 8; ++i) bm = fmaxf(bm, red[i]);
  __syncthreads();

  float sum = 0.f;
  for (int j = threadIdx.x; j < len; j += 256) {
    float e = __expf(s[j] - bm);
    a[j] = (bf16_t)e;
    sum += e;
  }
  int pad_end = (len + 31) & ~31;  // AV kernel reads up to the 32-wide K tile
  for (int j = len + (int)threadIdx.x; j < pad_end; j += 256) a[j] = (bf16_t)0.f;
  for (int o = 16; o > 0; o >>= 1) sum += __shfl_xor(sum, o, 32);
  if ((threadIdx.x & 31) == 0) red[threadIdx.x >> 5] = sum;
  __syncthreads();
  if (threadIdx.x == 0) {
    float t = 0.f;
    for (int i = 0; i < 8; ++i) t += red[i];
    rowsum[(size_t)bh * kT + qrow] = t;
  }
}

// ------------------------------------------- y = (att @ V) / rowsum -> [BT,C]
// per (b,h): A=att[T,T], Bt=vT[D,T]; block tile 128(M) x 64(N), causal K bound.
__global__ __launch_bounds__(256)
void attn_av_kernel(const bf16_t* __restrict__ att, const bf16_t* __restrict__ vT,
                    const float* __restrict__ rowsum, bf16_t* __restrict__ y) {
  int bh = blockIdx.z, b = bh / kH, h = bh % kH;
  const bf16_t* A  = att + (size_t)bh * kT * kT;
  const bf16_t* Bt = vT  + (size_t)bh * kD * kT;
  int lane = threadIdx.x & 31;
  int wave = threadIdx.x >> 5;
  int wm = wave >> 1, wn = wave & 1;
  int row0 = blockIdx.y * 128 + wm * 32;
  int col0 = wn * 32;

  v8f acc[2][2] = {};
  int kend = row0 + 32;  // causal: only k tiles with k0 <= row0+31
  for (int k0 = 0; k0 < kend; k0 += 32) {
    v16bf a0 = load_a_frag(A, kT, row0,      k0, lane);
    v16bf a1 = load_a_frag(A, kT, row0 + 16, k0, lane);
    v16bf b0 = load_b_frag(Bt, kT, col0,      k0, lane, kD - 1);
    v16bf b1 = load_b_frag(Bt, kT, col0 + 16, k0, lane, kD - 1);
    acc[0][0] = wmma_bf16(a0, b0, acc[0][0]);
    acc[0][1] = wmma_bf16(a0, b1, acc[0][1]);
    acc[1][0] = wmma_bf16(a1, b0, acc[1][0]);
    acc[1][1] = wmma_bf16(a1, b1, acc[1][1]);
  }
  int n = lane & 15, half = lane >> 4;
#pragma unroll
  for (int tm = 0; tm < 2; ++tm)
#pragma unroll
    for (int tn = 0; tn < 2; ++tn)
#pragma unroll
      for (int r = 0; r < 8; ++r) {
        int grow = row0 + tm * 16 + r + 8 * half;
        int gcol = col0 + tn * 16 + n;
        float rs = rowsum[(size_t)bh * kT + grow];
        float v = acc[tm][tn][r] / rs;
        y[((size_t)(b * kT + grow)) * kC + h * kD + gcol] = (bf16_t)v;
      }
}

// --------------------------------------------------------------- small kernels

__global__ __launch_bounds__(256)
void embed_kernel(const int* __restrict__ idx, const float* __restrict__ wte,
                  const float* __restrict__ wpe, float* __restrict__ x) {
  int i = blockIdx.x * 256 + threadIdx.x;
  if (i >= kBT * kC) return;
  int row = i / kC, c = i - row * kC;
  int t = row % kT;
  int tok = idx[row];
  x[i] = wte[(size_t)tok * kC + c] + wpe[(size_t)t * kC + c];
}

__global__ __launch_bounds__(256)
void layernorm_kernel(const float* __restrict__ x, const float* __restrict__ w,
                      const float* __restrict__ bp, bf16_t* __restrict__ out) {
  int row = blockIdx.x;
  const float* xr = x + (size_t)row * kC;
  __shared__ float rs[8], rss[8];
  float s = 0.f, ss = 0.f;
  for (int c = threadIdx.x; c < kC; c += 256) { float v = xr[c]; s += v; ss += v * v; }
  for (int o = 16; o > 0; o >>= 1) { s += __shfl_xor(s, o, 32); ss += __shfl_xor(ss, o, 32); }
  if ((threadIdx.x & 31) == 0) { rs[threadIdx.x >> 5] = s; rss[threadIdx.x >> 5] = ss; }
  __syncthreads();
  float ts = 0.f, tss = 0.f;
  for (int i = 0; i < 8; ++i) { ts += rs[i]; tss += rss[i]; }
  float mean = ts / kC;
  float var = tss / kC - mean * mean;
  float inv = rsqrtf(var + 1e-5f);
  for (int c = threadIdx.x; c < kC; c += 256)
    out[(size_t)row * kC + c] = (bf16_t)((xr[c] - mean) * inv * w[c] + bp[c]);
}

__global__ __launch_bounds__(256)
void split_qkv_kernel(const float* __restrict__ qkv, bf16_t* __restrict__ q,
                      bf16_t* __restrict__ k, bf16_t* __restrict__ vT) {
  int i = blockIdx.x * 256 + threadIdx.x;
  if (i >= kBT * kC) return;
  int row = i / kC, c = i - row * kC;
  int b = row / kT, t = row - b * kT;
  int h = c / kD, d = c - h * kD;
  size_t bh = (size_t)(b * kH + h);
  size_t src = (size_t)row * k3C + c;
  q [bh * kT * kD + (size_t)t * kD + d] = (bf16_t)qkv[src];
  k [bh * kT * kD + (size_t)t * kD + d] = (bf16_t)qkv[src + kC];
  vT[bh * kD * kT + (size_t)d * kT + t] = (bf16_t)qkv[src + 2 * kC];  // transposed
}

// f32 [L][K][N] -> bf16 [L][N][K]   (weights into WMMA B layout)
__global__ __launch_bounds__(256)
void transpose_convert_kernel(const float* __restrict__ in, bf16_t* __restrict__ out,
                              int K, int N) {
  __shared__ bf16_t tile[32][33];
  size_t base = (size_t)blockIdx.z * K * N;
  int kb = blockIdx.y * 32, nb = blockIdx.x * 32;
  for (int i = threadIdx.y; i < 32; i += 8)
    tile[i][threadIdx.x] = (bf16_t)in[base + (size_t)(kb + i) * N + nb + threadIdx.x];
  __syncthreads();
  for (int i = threadIdx.y; i < 32; i += 8)
    out[base + (size_t)(nb + i) * K + kb + threadIdx.x] = tile[threadIdx.x][i];
}

__global__ __launch_bounds__(256)
void convert_bf16_kernel(const float* __restrict__ in, bf16_t* __restrict__ out, size_t n) {
  size_t i = (size_t)blockIdx.x * 256 + threadIdx.x;
  if (i < n) out[i] = (bf16_t)in[i];
}

// ------------------------------------------------------------------- launcher

extern "C" void kernel_launch(void* const* d_in, const int* in_sizes, int n_in,
                              void* d_out, int out_size, void* d_ws, size_t ws_size,
                              hipStream_t stream) {
  const int*   idx    = (const int*)  d_in[0];
  const float* wte    = (const float*)d_in[1];
  const float* wpe    = (const float*)d_in[2];
  const float* ln1_w  = (const float*)d_in[3];
  const float* ln1_b  = (const float*)d_in[4];
  const float* qkv_w  = (const float*)d_in[5];
  const float* qkv_b  = (const float*)d_in[6];
  const float* proj_w = (const float*)d_in[7];
  const float* proj_b = (const float*)d_in[8];
  const float* ln2_w  = (const float*)d_in[9];
  const float* ln2_b  = (const float*)d_in[10];
  const float* fc_w   = (const float*)d_in[11];
  const float* fc_b   = (const float*)d_in[12];
  const float* fco_w  = (const float*)d_in[13];
  const float* fco_b  = (const float*)d_in[14];
  const float* lnf_w  = (const float*)d_in[15];
  const float* lnf_b  = (const float*)d_in[16];

  char* base = (char*)d_ws;
  size_t off = 0;
  auto alloc = [&](size_t bytes) -> char* {
    char* p = base + off;
    off = (off + bytes + 255) & ~(size_t)255;
    return p;
  };
  float*  x      = (float*) alloc((size_t)kBT * kC * 4);
  bf16_t* actC   = (bf16_t*)alloc((size_t)kBT * kC * 2);
  bf16_t* act4C  = (bf16_t*)alloc((size_t)kBT * k4C * 2);
  float*  qkvf   = (float*) alloc((size_t)kBT * k3C * 4);
  bf16_t* qb     = (bf16_t*)alloc((size_t)kBH * kT * kD * 2);
  bf16_t* kb     = (bf16_t*)alloc((size_t)kBH * kT * kD * 2);
  bf16_t* vTb    = (bf16_t*)alloc((size_t)kBH * kD * kT * 2);
  float*  scores = (float*) alloc((size_t)kBH * kT * kT * 4);
  bf16_t* att    = (bf16_t*)alloc((size_t)kBH * kT * kT * 2);
  float*  rowsum = (float*) alloc((size_t)kBH * kT * 4);
  bf16_t* wte_bf = (bf16_t*)alloc((size_t)kV * kC * 2);
  bf16_t* qkvwT  = (bf16_t*)alloc((size_t)kL * kC * k3C * 2);
  bf16_t* projwT = (bf16_t*)alloc((size_t)kL * kC * kC * 2);
  bf16_t* fcwT   = (bf16_t*)alloc((size_t)kL * kC * k4C * 2);
  bf16_t* fcowT  = (bf16_t*)alloc((size_t)kL * k4C * kC * 2);

  // ---- weight precision/layout conversion (bf16, B transposed to [N,K]) ----
  {
    size_t n = (size_t)kV * kC;
    convert_bf16_kernel<<<(unsigned)((n + 255) / 256), 256, 0, stream>>>(wte, wte_bf, n);
  }
  dim3 tb(32, 8);
  transpose_convert_kernel<<<dim3(k3C / 32, kC / 32, kL),  tb, 0, stream>>>(qkv_w,  qkvwT,  kC,  k3C);
  transpose_convert_kernel<<<dim3(kC / 32,  kC / 32, kL),  tb, 0, stream>>>(proj_w, projwT, kC,  kC);
  transpose_convert_kernel<<<dim3(k4C / 32, kC / 32, kL),  tb, 0, stream>>>(fc_w,   fcwT,   kC,  k4C);
  transpose_convert_kernel<<<dim3(kC / 32,  k4C / 32, kL), tb, 0, stream>>>(fco_w,  fcowT,  k4C, kC);

  embed_kernel<<<(kBT * kC) / 256, 256, 0, stream>>>(idx, wte, wpe, x);

  const float scale = 0.125f;  // 1/sqrt(64)
  for (int l = 0; l < kL; ++l) {
    layernorm_kernel<<<kBT, 256, 0, stream>>>(x, ln1_w + l * kC, ln1_b + l * kC, actC);

    gemm_bf16_kernel<<<dim3(k3C / 128, kBT / 64), 256, 0, stream>>>(
        actC, kC, qkvwT + (size_t)l * k3C * kC, kC, qkv_b + l * k3C,
        nullptr, qkvf, nullptr, kBT, k3C, kC, k3C, MODE_F32);

    split_qkv_kernel<<<(kBT * kC) / 256, 256, 0, stream>>>(qkvf, qb, kb, vTb);

    attn_score_kernel<<<dim3(kT / 128, kT / 64, kBH), 256, 0, stream>>>(qb, kb, scores, scale);
    softmax_kernel<<<dim3(kT, kBH), 256, 0, stream>>>(scores, att, rowsum);
    attn_av_kernel<<<dim3(1, kT / 128, kBH), 256, 0, stream>>>(att, vTb, rowsum, actC);

    gemm_bf16_kernel<<<dim3(kC / 128, kBT / 64), 256, 0, stream>>>(
        actC, kC, projwT + (size_t)l * kC * kC, kC, proj_b + l * kC,
        x, x, nullptr, kBT, kC, kC, kC, MODE_RESID);

    layernorm_kernel<<<kBT, 256, 0, stream>>>(x, ln2_w + l * kC, ln2_b + l * kC, actC);

    gemm_bf16_kernel<<<dim3(k4C / 128, kBT / 64), 256, 0, stream>>>(
        actC, kC, fcwT + (size_t)l * k4C * kC, kC, fc_b + l * k4C,
        nullptr, nullptr, act4C, kBT, k4C, kC, k4C, MODE_GELU);

    gemm_bf16_kernel<<<dim3(kC / 128, kBT / 64), 256, 0, stream>>>(
        act4C, k4C, fcowT + (size_t)l * kC * k4C, k4C, fco_b + l * kC,
        x, x, nullptr, kBT, kC, k4C, kC, MODE_RESID);
  }

  layernorm_kernel<<<kBT, 256, 0, stream>>>(x, lnf_w, lnf_b, actC);

  // logits = lnf(x) @ wte^T : wte[V,C] row-major is already the [N,K] B layout
  gemm_bf16_kernel<<<dim3((kV + 127) / 128, kBT / 64), 256, 0, stream>>>(
      actC, kC, wte_bf, kC, nullptr, nullptr, (float*)d_out, nullptr,
      kBT, kV, kC, kV, MODE_F32);
}
